// ModernMLP_75041668596080
// MI455X (gfx1250) — matrix operand order repaired and checked by reading, not compiled, not run on hardware
//
#include <hip/hip_runtime.h>
#include <math.h>

// ---------------- types ----------------
typedef __attribute__((ext_vector_type(16))) __bf16 v16bf;
typedef __attribute__((ext_vector_type(8)))  float  v8f;

struct U4 { unsigned int x, y, z, w; };
struct F4 { float x, y, z, w; };

union AFrag {                 // 16 bf16 = 8 VGPRs
    v16bf v;
    unsigned short u[16];
    unsigned int   w32[8];
    U4 q[2];
};

__device__ __forceinline__ unsigned short f2bf(float f) {
    unsigned int u = __float_as_uint(f);
    unsigned int r = (u + 0x7FFFu + ((u >> 16) & 1u)) >> 16;   // RNE
    return (unsigned short)r;
}

#if __has_builtin(__builtin_amdgcn_cvt_pk_bf16_f32)
typedef __attribute__((ext_vector_type(2))) __bf16 v2bf;
__device__ __forceinline__ unsigned int pk2bf(float lo, float hi) {
    union { v2bf v; unsigned int u; } cvt;
    cvt.v = __builtin_amdgcn_cvt_pk_bf16_f32(lo, hi);          // native RNE pack
    return cvt.u;
}
#else
__device__ __forceinline__ unsigned int pk2bf(float lo, float hi) {
    return (unsigned int)f2bf(lo) | ((unsigned int)f2bf(hi) << 16);
}
#endif

__device__ __forceinline__ float bf2f(unsigned short b) {
    return __uint_as_float((unsigned int)b << 16);
}
__device__ __forceinline__ float silu(float v) { return v / (1.f + __expf(-v)); }

static constexpr int   Nrows = 4096;
static constexpr int   Dd    = 2048;
static constexpr int   Hh    = 8192;
static constexpr int   Rr    = 16;
static constexpr int   DEPTH = 2;

static constexpr int   BM    = 256;   // block rows
static constexpr int   BN    = 64;    // block cols
static constexpr int   BK    = 64;    // K-slab per stage
static constexpr int   LDA   = 72;    // padded LDS row stride (elements, 144B = 9*16B)
static constexpr int   RB    = Nrows / BM;   // 16 row-blocks

// ---------------- rmsnorm (+ optional router) ----------------
template <bool ROUTER>
__global__ void rmsnorm_kernel(const float* __restrict__ X,
                               const float* __restrict__ nw,
                               const float* __restrict__ rw,
                               unsigned short* __restrict__ Xbf,
                               int* __restrict__ active, int D) {
    int row = blockIdx.x, t = threadIdx.x;
    const float* xr = X + (size_t)row * D;
    float v[8]; float ss = 0.f;
#pragma unroll
    for (int k = 0; k < 8; ++k) { v[k] = xr[t + 256 * k]; ss += v[k] * v[k]; }
    __shared__ float sh[256];
    sh[t] = ss; __syncthreads();
    for (int s = 128; s > 0; s >>= 1) { if (t < s) sh[t] += sh[t + s]; __syncthreads(); }
    float factor = rsqrtf(sh[0] / (float)D + 1e-6f);
    __syncthreads();
    float dot = 0.f;
#pragma unroll
    for (int k = 0; k < 8; ++k) {
        int c = t + 256 * k;
        float xn = v[k] * factor * nw[c];
        Xbf[(size_t)row * D + c] = f2bf(xn);
        if (ROUTER) dot += xn * rw[c];
    }
    if (ROUTER) {
        sh[t] = dot; __syncthreads();
        for (int s = 128; s > 0; s >>= 1) { if (t < s) sh[t] += sh[t + s]; __syncthreads(); }
        if (t == 0) {
            float g = 1.f / (1.f + expf(-sh[0]));
            active[row] = (g > 0.35f) ? 1 : 0;
        }
    }
}

// ---------------- LDS-tiled GEMM core pieces ----------------
// Stage A: 256 rows x 32 K of bf16 -> LDS (one row per thread, 4x16B)
__device__ __forceinline__ void stage_a(unsigned short* ldsA,
                                        const unsigned short* ag, int t) {
    const U4* src = (const U4*)ag;
    U4* dst = (U4*)(ldsA + t * LDA);
    dst[0] = src[0]; dst[1] = src[1]; dst[2] = src[2]; dst[3] = src[3];
    __builtin_prefetch(ag + BK, 0, 1);          // next K-slab (speculative)
}

// Stage B: 64 weight rows x 64 K fp32 -> bf16 in LDS (16 floats per thread)
template <bool QUANT>
__device__ __forceinline__ void stage_b(unsigned short* ldsB,
                                        const float* wg, int bcol, int bchunk,
                                        float qs, float qis) {
    const F4* src = (const F4*)wg;
    unsigned int* dst = (unsigned int*)(ldsB + bcol * LDA + bchunk * 16);
#pragma unroll
    for (int i = 0; i < 4; ++i) {
        F4 f = src[i];
        float e0 = f.x, e1 = f.y, e2 = f.z, e3 = f.w;
        if (QUANT) {
            e0 = fminf(1.f, fmaxf(-1.f, rintf(e0 * qis))) * qs;
            e1 = fminf(1.f, fmaxf(-1.f, rintf(e1 * qis))) * qs;
            e2 = fminf(1.f, fmaxf(-1.f, rintf(e2 * qis))) * qs;
            e3 = fminf(1.f, fmaxf(-1.f, rintf(e3 * qis))) * qs;
        }
        dst[2 * i + 0] = pk2bf(e0, e1);
        dst[2 * i + 1] = pk2bf(e2, e3);
    }
    __builtin_prefetch(wg + BK, 0, 1);          // next K-slab (speculative)
}

// ---------------- up-projection GEMM (LDS tiled): Out_bf16 = [silu](A @ W^T) ----------------
// A: [N,K] bf16 ; W: [Ncols,K] f32 ; Out: [N,Ncols] bf16
template <bool QUANT, bool SILU>
__global__ void __launch_bounds__(256)
gemm_up_lds(const unsigned short* __restrict__ Abf,
            const float* __restrict__ W,
            const float* __restrict__ scale_sum, float inv_count,
            unsigned short* __restrict__ Out,
            int K, int Ncols) {
    __shared__ unsigned short ldsA[BM * LDA];
    __shared__ unsigned short ldsB[BN * LDA];

    int t = threadIdx.x;
    int bm = blockIdx.x & (RB - 1);
    int bn = blockIdx.x >> 4;                   // RB == 16
    int rm = bm * BM, cn = bn * BN;
    int w = t >> 5, lane = t & 31, m = lane & 15, half = lane >> 4;
    int wm = w & 3, wn = w >> 2;                // 4 M-groups x 2 N-groups
    int bcol = t >> 2, bchunk = t & 3;

    float qs = 1.f, qis = 1.f;
    if (QUANT) { float s = fmaxf(scale_sum[0] * inv_count, 1e-8f); qs = s; qis = 1.f / s; }

    const unsigned short* ag = Abf + (size_t)(rm + t) * K;
    const float* wg = W + (size_t)(cn + bcol) * K + bchunk * 16;

    v8f acc[4][2] = {};
    for (int k0 = 0; k0 < K; k0 += BK) {
        stage_a(ldsA, ag + k0, t);
        stage_b<QUANT>(ldsB, wg + k0, bcol, bchunk, qs, qis);
        __syncthreads();
#pragma unroll
        for (int ks = 0; ks < BK; ks += 32) {
            AFrag b[2];
#pragma unroll
            for (int j = 0; j < 2; ++j) {
                const unsigned short* bp = ldsB + (wn * 32 + j * 16 + m) * LDA + ks + half * 16;
                b[j].q[0] = ((const U4*)bp)[0];
                b[j].q[1] = ((const U4*)bp)[1];
            }
#pragma unroll
            for (int mi = 0; mi < 4; ++mi) {
                const unsigned short* ap = ldsA + (wm * 64 + mi * 16 + m) * LDA + ks;
                AFrag a;
                a.q[0] = *(const U4*)(ap + half * 16);
                a.q[1] = *(const U4*)(ap + half * 16 + 8);
#pragma unroll
                for (int j = 0; j < 2; ++j)
                    acc[mi][j] = __builtin_amdgcn_wmma_f32_16x16x32_bf16(
                        false, a.v, false, b[j].v, (short)0, acc[mi][j], false, false);
            }
        }
        __syncthreads();
    }
    // epilogue: row = rm + wm*64 + mi*16 + half*8 + r ; col = cn + wn*32 + j*16 + m
#pragma unroll
    for (int mi = 0; mi < 4; ++mi)
#pragma unroll
        for (int j = 0; j < 2; ++j)
#pragma unroll
            for (int r = 0; r < 8; ++r) {
                int row = rm + wm * 64 + mi * 16 + half * 8 + r;
                int col = cn + wn * 32 + j * 16 + m;
                float val = acc[mi][j][r];
                if (SILU) val = silu(val);
                Out[(size_t)row * Ncols + col] = f2bf(val);
            }
}

// ---------------- down-projection GEMM (LDS tiled): X += (A @ W^T) * gamma [* active] ----------------
template <bool MASK, bool QUANT>
__global__ void __launch_bounds__(256)
gemm_down_lds(const unsigned short* __restrict__ Abf,
              const float* __restrict__ W,
              const float* __restrict__ gamma,
              const int* __restrict__ active,
              const float* __restrict__ scale_sum, float inv_count,
              float* __restrict__ X,
              int K, int Dcols) {
    __shared__ unsigned short ldsA[BM * LDA];
    __shared__ unsigned short ldsB[BN * LDA];

    int t = threadIdx.x;
    int bm = blockIdx.x & (RB - 1);
    int bn = blockIdx.x >> 4;
    int rm = bm * BM, cn = bn * BN;
    int w = t >> 5, lane = t & 31, m = lane & 15, half = lane >> 4;
    int wm = w & 3, wn = w >> 2;
    int bcol = t >> 2, bchunk = t & 3;

    float qs = 1.f, qis = 1.f;
    if (QUANT) { float s = fmaxf(scale_sum[0] * inv_count, 1e-8f); qs = s; qis = 1.f / s; }

    const unsigned short* ag = Abf + (size_t)(rm + t) * K;
    const float* wg = W + (size_t)(cn + bcol) * K + bchunk * 16;

    v8f acc[4][2] = {};
    for (int k0 = 0; k0 < K; k0 += BK) {
        stage_a(ldsA, ag + k0, t);
        stage_b<QUANT>(ldsB, wg + k0, bcol, bchunk, qs, qis);
        __syncthreads();
#pragma unroll
        for (int ks = 0; ks < BK; ks += 32) {
            AFrag b[2];
#pragma unroll
            for (int j = 0; j < 2; ++j) {
                const unsigned short* bp = ldsB + (wn * 32 + j * 16 + m) * LDA + ks + half * 16;
                b[j].q[0] = ((const U4*)bp)[0];
                b[j].q[1] = ((const U4*)bp)[1];
            }
#pragma unroll
            for (int mi = 0; mi < 4; ++mi) {
                const unsigned short* ap = ldsA + (wm * 64 + mi * 16 + m) * LDA + ks;
                AFrag a;
                a.q[0] = *(const U4*)(ap + half * 16);
                a.q[1] = *(const U4*)(ap + half * 16 + 8);
#pragma unroll
                for (int j = 0; j < 2; ++j)
                    acc[mi][j] = __builtin_amdgcn_wmma_f32_16x16x32_bf16(
                        false, a.v, false, b[j].v, (short)0, acc[mi][j], false, false);
            }
        }
        __syncthreads();
    }
#pragma unroll
    for (int mi = 0; mi < 4; ++mi)
#pragma unroll
        for (int j = 0; j < 2; ++j)
#pragma unroll
            for (int r = 0; r < 8; ++r) {
                int row = rm + wm * 64 + mi * 16 + half * 8 + r;
                int col = cn + wn * 32 + j * 16 + m;
                float add = acc[mi][j][r] * gamma[col];
                if (MASK) add = active[row] ? add : 0.f;
                X[(size_t)row * Dcols + col] += add;
            }
}

// ---------------- SwiGLU combine: h1 = silu(h1) * h3 (bf16, in place on h1) ----------------
__global__ void swiglu_combine(unsigned short* __restrict__ h1,
                               const unsigned short* __restrict__ h3,
                               long long n) {
    long long i = (long long)blockIdx.x * blockDim.x + threadIdx.x;
    long long stride = (long long)gridDim.x * blockDim.x;
    for (; i < n; i += stride) {
        float u = bf2f(h1[i]);
        float g = bf2f(h3[i]);
        h1[i] = f2bf(silu(u) * g);
    }
}

// ---------------- AdaptiveRankLinear (rank 16) ----------------
__global__ void arl_scale(const float* __restrict__ S, float* __restrict__ s_eff, int R) {
    if (threadIdx.x == 0 && blockIdx.x == 0) {
        float tot = 0.f;
        for (int i = 0; i < R; ++i) tot += fabsf(S[i]);
        tot = fmaxf(tot, 1e-8f);
        float cum = 0.f;
        for (int i = 0; i < R; ++i) {
            bool keep = (i == 0) || (cum / tot < 0.95f);
            s_eff[i] = keep ? S[i] : 0.f;
            cum += fabsf(S[i]);
        }
    }
}

__global__ void arl_low(const float* __restrict__ X, const float* __restrict__ nw,
                        const float* __restrict__ U, const float* __restrict__ s_eff,
                        float* __restrict__ low, int D) {
    int row = blockIdx.x, t = threadIdx.x;
    const float* xr = X + (size_t)row * D;
    float v[8]; float ss = 0.f;
#pragma unroll
    for (int k = 0; k < 8; ++k) { v[k] = xr[t + 256 * k]; ss += v[k] * v[k]; }
    __shared__ float sh[256];
    sh[t] = ss; __syncthreads();
    for (int s = 128; s > 0; s >>= 1) { if (t < s) sh[t] += sh[t + s]; __syncthreads(); }
    float factor = rsqrtf(sh[0] / (float)D + 1e-6f);
    __syncthreads();
    float p[16];
#pragma unroll
    for (int r = 0; r < 16; ++r) p[r] = 0.f;
#pragma unroll
    for (int k = 0; k < 8; ++k) {
        int c = t + 256 * k;
        float xn = v[k] * factor * nw[c];
        const float* Urow = U + (size_t)c * 16;
#pragma unroll
        for (int r = 0; r < 16; ++r) p[r] += xn * Urow[r];
    }
    __shared__ float sp[256 * 16];
#pragma unroll
    for (int r = 0; r < 16; ++r) sp[t * 16 + r] = p[r];
    __syncthreads();
    for (int s = 128; s > 0; s >>= 1) {
        if (t < s)
            for (int r = 0; r < 16; ++r) sp[t * 16 + r] += sp[(t + s) * 16 + r];
        __syncthreads();
    }
    if (t < 16) low[(size_t)row * 16 + t] = sp[t] * s_eff[t];
}

__global__ void arl_addv(const float* __restrict__ low, const float* __restrict__ V,
                         const float* __restrict__ gamma, float* __restrict__ X, int D) {
    int row = blockIdx.x, t = threadIdx.x;
    __shared__ float l[16];
    if (t < 16) l[t] = low[(size_t)row * 16 + t];
    __syncthreads();
#pragma unroll
    for (int k = 0; k < 8; ++k) {
        int c = t + 256 * k;
        float acc = 0.f;
#pragma unroll
        for (int r = 0; r < 16; ++r) acc += l[r] * V[(size_t)r * D + c];
        X[(size_t)row * D + c] += acc * gamma[c];
    }
}

// ---------------- absmean reduction for ternary scale ----------------
__global__ void absmean_partial(const float* __restrict__ W, long long n,
                                float* __restrict__ partial) {
    long long i = (long long)blockIdx.x * 256 + threadIdx.x;
    long long stride = (long long)gridDim.x * 256;
    float s = 0.f;
    for (; i < n; i += stride) s += fabsf(W[i]);
    __shared__ float sh[256];
    sh[threadIdx.x] = s; __syncthreads();
    for (int k = 128; k > 0; k >>= 1) { if (threadIdx.x < k) sh[threadIdx.x] += sh[threadIdx.x + k]; __syncthreads(); }
    if (threadIdx.x == 0) partial[blockIdx.x] = sh[0];
}

__global__ void absmean_final(const float* __restrict__ partial, int n, float* __restrict__ out) {
    int t = threadIdx.x; float s = 0.f;
    for (int i = t; i < n; i += 256) s += partial[i];
    __shared__ float sh[256];
    sh[t] = s; __syncthreads();
    for (int k = 128; k > 0; k >>= 1) { if (t < k) sh[t] += sh[t + k]; __syncthreads(); }
    if (t == 0) out[0] = sh[0];
}

// ---------------- launcher ----------------
extern "C" void kernel_launch(void* const* d_in, const int* in_sizes, int n_in,
                              void* d_out, int out_size, void* d_ws, size_t ws_size,
                              hipStream_t stream) {
    (void)in_sizes; (void)n_in; (void)out_size; (void)ws_size;
    const float* x_in      = (const float*)d_in[0];
    const float* acb_norm  = (const float*)d_in[1];
    const float* router_w  = (const float*)d_in[2];
    const float* w1        = (const float*)d_in[3];
    const float* w2        = (const float*)d_in[4];
    const float* w3        = (const float*)d_in[5];
    const float* acb_gamma = (const float*)d_in[6];
    const float* arl_norm  = (const float*)d_in[7];
    const float* U         = (const float*)d_in[8];
    const float* S         = (const float*)d_in[9];
    const float* V         = (const float*)d_in[10];
    const float* arl_gamma = (const float*)d_in[11];
    const float* bit_norm  = (const float*)d_in[12];
    const float* w_up      = (const float*)d_in[13];
    const float* w_down    = (const float*)d_in[14];
    const float* bit_gamma = (const float*)d_in[15];

    float* X = (float*)d_out;

    // workspace carve-up
    char* ws = (char*)d_ws;
    unsigned short* xn_bf = (unsigned short*)ws;                    // N*D*2  = 16 MB
    ws += (size_t)Nrows * Dd * 2;
    unsigned short* h1 = (unsigned short*)ws;                       // N*H*2  = 64 MB
    ws += (size_t)Nrows * Hh * 2;
    unsigned short* h3 = (unsigned short*)ws;                       // N*H*2  = 64 MB
    ws += (size_t)Nrows * Hh * 2;
    int*   active = (int*)ws;    ws += (size_t)Nrows * 4;
    float* low    = (float*)ws;  ws += (size_t)Nrows * Rr * 4;
    float* s_eff  = (float*)ws;  ws += 64;
    float* partA  = (float*)ws;  ws += 1024 * 4;
    float* partB  = (float*)ws;  ws += 1024 * 4;
    float* sumA   = (float*)ws;  ws += 16;
    float* sumB   = (float*)ws;  ws += 16;

    hipMemcpyAsync(X, x_in, (size_t)Nrows * Dd * sizeof(float),
                   hipMemcpyDeviceToDevice, stream);

    const int upBlocks   = RB * (Hh / BN);     // 16 * 128 = 2048
    const int downBlocks = RB * (Dd / BN);     // 16 * 32  = 512
    const float inv_cnt  = 1.f / ((float)Hh * (float)Dd);

    for (int d = 0; d < DEPTH; ++d) {
        const size_t wOff = (size_t)d * Hh * Dd;
        const size_t vOff = (size_t)d * Dd;

        // ---- AdaptiveComputeBlock ----
        rmsnorm_kernel<true><<<Nrows, 256, 0, stream>>>(
            X, acb_norm + vOff, router_w + vOff, xn_bf, active, Dd);
        gemm_up_lds<false, false><<<upBlocks, 256, 0, stream>>>(
            xn_bf, w1 + wOff, nullptr, 0.f, h1, Dd, Hh);
        gemm_up_lds<false, false><<<upBlocks, 256, 0, stream>>>(
            xn_bf, w3 + wOff, nullptr, 0.f, h3, Dd, Hh);
        swiglu_combine<<<4096, 256, 0, stream>>>(
            h1, h3, (long long)Nrows * Hh);
        gemm_down_lds<true, false><<<downBlocks, 256, 0, stream>>>(
            h1, w2 + wOff, acb_gamma + vOff, active, nullptr, 0.f, X, Hh, Dd);

        // ---- AdaptiveRankLinear ----
        arl_scale<<<1, 32, 0, stream>>>(S + (size_t)d * Rr, s_eff, Rr);
        arl_low<<<Nrows, 256, 0, stream>>>(
            X, arl_norm + vOff, U + (size_t)d * Dd * Rr, s_eff, low, Dd);
        arl_addv<<<Nrows, 256, 0, stream>>>(
            low, V + (size_t)d * Rr * Dd, arl_gamma + vOff, X, Dd);

        // ---- BitLinear ----
        rmsnorm_kernel<false><<<Nrows, 256, 0, stream>>>(
            X, bit_norm + vOff, nullptr, xn_bf, nullptr, Dd);
        absmean_partial<<<1024, 256, 0, stream>>>(
            w_up + wOff, (long long)Hh * Dd, partA);
        absmean_final<<<1, 256, 0, stream>>>(partA, 1024, sumA);
        absmean_partial<<<1024, 256, 0, stream>>>(
            w_down + wOff, (long long)Hh * Dd, partB);
        absmean_final<<<1, 256, 0, stream>>>(partB, 1024, sumB);
        gemm_up_lds<true, true><<<upBlocks, 256, 0, stream>>>(
            xn_bf, w_up + wOff, sumA, inv_cnt, h1, Dd, Hh);
        gemm_down_lds<false, true><<<downBlocks, 256, 0, stream>>>(
            h1, w_down + wOff, bit_gamma + vOff, nullptr, sumB, inv_cnt, X, Hh, Dd);
    }
}
